// MeshEdgeUpdate_21998822490255
// MI455X (gfx1250) — matrix operand order repaired and checked by reading, not compiled, not run on hardware
//
#include <hip/hip_runtime.h>
#include <hip/hip_bf16.h>

// ---------------- problem constants ----------------
static constexpr long GN_  = 65160;
static constexpr long MN_  = 40962;
static constexpr long ME_  = 327660;
static constexpr long GE_  = 163848;
static constexpr long EMB  = 256;     // MEMB == EEMB
static constexpr int  XDIM = 768;     // 2*MEMB + EEMB
static constexpr int  H1D  = 512;
static constexpr int  H2D  = 64;

// ---------------- WMMA vector types ----------------
typedef __attribute__((ext_vector_type(16))) __bf16          v16bf;
typedef __attribute__((ext_vector_type(8)))  float           v8f;
typedef __attribute__((ext_vector_type(16))) unsigned short  v16us;
typedef __attribute__((ext_vector_type(8)))  unsigned int    v8u;

__device__ __forceinline__ unsigned short f2bf(float f) {
  unsigned u = __builtin_bit_cast(unsigned, f);
  u += 0x7FFFu + ((u >> 16) & 1u);           // round-to-nearest-even
  return (unsigned short)(u >> 16);
}
__device__ __forceinline__ float silu(float x) { return x / (1.f + __expf(-x)); }

// A-fragment (16xK bf16, M in lanes) per ISA 16-bit A layout:
// lane l: row m=l&15 ; dword v: k = (v<4?0:16) + (l>>4)*8 + (v&3)*2 + {0,1}
__device__ __forceinline__ v16bf load_a_frag(const unsigned short* lds, int stride,
                                             int m, int lh, int kbase) {
  v16us a;
#pragma unroll
  for (int v = 0; v < 8; ++v) {
    int k0 = ((v < 4) ? 0 : 16) + lh * 8 + (v & 3) * 2;
    unsigned d = *(const unsigned*)(lds + m * stride + kbase + k0);
    a[2 * v]     = (unsigned short)(d & 0xffffu);
    a[2 * v + 1] = (unsigned short)(d >> 16);
  }
  return __builtin_bit_cast(v16bf, a);
}

// B-fragment from pre-swizzled weights: one contiguous 32B per lane.
__device__ __forceinline__ v16bf load_b_frag(const unsigned short* __restrict__ w,
                                             long blk, int lane) {
  v8u d = *(const v8u*)(w + (blk * 32 + lane) * 16);
  return __builtin_bit_cast(v16bf, d);
}

// ---------------- kernels ----------------
__global__ void init_accum_kernel(double* accum) { accum[0] = 0.0; accum[1] = 0.0; }

// swizzle W[K,N] f32 -> bf16 WMMA-B layout; flat: (((kt*NT+nt)*32+lane)*16+slot)
// maps element k = kt*32 + (slot>>3)*16 + (lane>>4)*8 + (slot&7) ; n = nt*16 + (lane&15)
__global__ void convw_kernel(const float* __restrict__ W, unsigned short* __restrict__ Ws,
                             int K, int N) {
  int NT = N >> 4;
  long idx = (long)blockIdx.x * blockDim.x + threadIdx.x;
  if (idx >= (long)K * N) return;
  int  slot = (int)(idx & 15);
  int  lane = (int)((idx >> 4) & 31);
  long bi   = idx >> 9;
  int  nt   = (int)(bi % NT);
  int  kt   = (int)(bi / NT);
  int  k = kt * 32 + (slot >> 3) * 16 + (lane >> 4) * 8 + (slot & 7);
  int  n = nt * 16 + (lane & 15);
  Ws[idx] = f2bf(W[(long)k * N + n]);
}

#define XS_STRIDE 772
#define H1_STRIDE 516
#define H2_STRIDE 68

__global__ __launch_bounds__(128) void mlp_kernel(
    const float* __restrict__ mx, const int* __restrict__ me_i,
    const float* __restrict__ me_x,
    const unsigned short* __restrict__ w1s, const unsigned short* __restrict__ w2s,
    const unsigned short* __restrict__ w3s,
    const float* __restrict__ b1, const float* __restrict__ b2,
    const float* __restrict__ b3,
    float* __restrict__ out_h, double* __restrict__ accum) {
  __shared__ unsigned short xs[16 * XS_STRIDE];
  __shared__ unsigned short h1s[16 * H1_STRIDE];
  __shared__ unsigned short h2s[16 * H2_STRIDE];
  __shared__ int   rows[16], cols[16];
  __shared__ float blk_sum[2];

  const int  tid  = threadIdx.x;
  const int  lane = tid & 31, wave = tid >> 5;
  const int  l15  = lane & 15, lh = lane >> 4;
  const long tbase = (long)blockIdx.x * 16;

  if (tid < 16) {
    long e = tbase + tid; if (e >= ME_) e = ME_ - 1;
    rows[tid] = me_i[e];
    cols[tid] = me_i[ME_ + e];
  }
  if (tid == 0) { blk_sum[0] = 0.f; blk_sum[1] = 0.f; }
  __syncthreads();

  // ---- stage X = [mx[row] | mx[col] | me_x]  (16 x 768) as bf16 in LDS ----
  for (int i = tid; i < 16 * (XDIM / 4); i += 128) {
    int e  = i / (XDIM / 4);
    int c4 = (i % (XDIM / 4)) * 4;
    long esrc = tbase + e; if (esrc >= ME_) esrc = ME_ - 1;
    const float* src;
    if (c4 < 256)      src = mx + (long)rows[e] * EMB + c4;
    else if (c4 < 512) src = mx + (long)cols[e] * EMB + (c4 - 256);
    else               src = me_x + esrc * EMB + (c4 - 512);
    float4 v = *(const float4*)src;
    unsigned short* dst = &xs[e * XS_STRIDE + c4];
    dst[0] = f2bf(v.x); dst[1] = f2bf(v.y); dst[2] = f2bf(v.z); dst[3] = f2bf(v.w);
  }
  __syncthreads();

  // ---- GEMM1: [16x768] @ [768x512], each wave owns 8 N-tiles ----
  v8f acc1[8] = {};
  for (int kt = 0; kt < XDIM / 32; ++kt) {
    v16bf a = load_a_frag(xs, XS_STRIDE, l15, lh, kt * 32);
#pragma unroll
    for (int j = 0; j < 8; ++j) {
      int nt = wave * 8 + j;
      v16bf b = load_b_frag(w1s, (long)kt * 32 + nt, lane);
      acc1[j] = __builtin_amdgcn_wmma_f32_16x16x32_bf16(
          false, a, false, b, (short)0, acc1[j], false, false);
    }
  }
#pragma unroll
  for (int j = 0; j < 8; ++j) {
    int nt  = wave * 8 + j;
    int col = nt * 16 + l15;
    float bias = b1[col];
#pragma unroll
    for (int r = 0; r < 8; ++r) {
      int row = r + 8 * lh;
      h1s[row * H1_STRIDE + col] = f2bf(silu(acc1[j][r] + bias));
    }
  }
  __syncthreads();

  // ---- GEMM2: [16x512] @ [512x64], one N-tile per wave ----
  v8f acc2 = {};
  for (int kt = 0; kt < H1D / 32; ++kt) {
    v16bf a = load_a_frag(h1s, H1_STRIDE, l15, lh, kt * 32);
    v16bf b = load_b_frag(w2s, (long)kt * 4 + wave, lane);
    acc2 = __builtin_amdgcn_wmma_f32_16x16x32_bf16(
        false, a, false, b, (short)0, acc2, false, false);
  }
  {
    int col = wave * 16 + l15;
    float bias = b2[col];
#pragma unroll
    for (int r = 0; r < 8; ++r) {
      int row = r + 8 * lh;
      h2s[row * H2_STRIDE + col] = f2bf(silu(acc2[r] + bias));
    }
  }
  __syncthreads();

  // ---- GEMM3: [16x64] @ [64x256], 4 N-tiles per wave; store pre-norm h ----
  float lsum = 0.f, lsq = 0.f;
#pragma unroll
  for (int j = 0; j < 4; ++j) {
    int nt = wave * 4 + j;
    v8f acc3 = {};
#pragma unroll
    for (int kt = 0; kt < 2; ++kt) {
      v16bf a = load_a_frag(h2s, H2_STRIDE, l15, lh, kt * 32);
      v16bf b = load_b_frag(w3s, (long)kt * 16 + nt, lane);
      acc3 = __builtin_amdgcn_wmma_f32_16x16x32_bf16(
          false, a, false, b, (short)0, acc3, false, false);
    }
    int col = nt * 16 + l15;
    float bias = b3[col];
#pragma unroll
    for (int r = 0; r < 8; ++r) {
      int  row = r + 8 * lh;
      long e   = tbase + row;
      if (e < ME_) {
        float x = acc3[r] + bias;
        out_h[e * EMB + col] = x;
        lsum += x; lsq += x * x;
      }
    }
  }
  atomicAdd(&blk_sum[0], lsum);
  atomicAdd(&blk_sum[1], lsq);
  __syncthreads();
  if (tid == 0) {
    atomicAdd(&accum[0], (double)blk_sum[0]);
    atomicAdd(&accum[1], (double)blk_sum[1]);
  }
}

__global__ __launch_bounds__(256) void norm_kernel(
    float* __restrict__ out_h, const float* __restrict__ me_x,
    const float* __restrict__ ln_w, const float* __restrict__ ln_b,
    const double* __restrict__ accum) {
  const double N = (double)ME_ * (double)EMB;
  double mean = accum[0] / N;
  double var  = accum[1] / N - mean * mean;
  float fm   = (float)mean;
  float rstd = (float)(1.0 / sqrt(var + 1e-5));
  long idx = ((long)blockIdx.x * blockDim.x + threadIdx.x) * 4;
  if (idx < ME_ * EMB) {
    float4 h  = *(float4*)(out_h + idx);
    float4 w  = *(const float4*)(ln_w + idx);
    float4 bb = *(const float4*)(ln_b + idx);
    float4 mexv = *(const float4*)(me_x + idx);
    h.x = (h.x - fm) * rstd * w.x + bb.x + mexv.x;
    h.y = (h.y - fm) * rstd * w.y + bb.y + mexv.y;
    h.z = (h.z - fm) * rstd * w.z + bb.z + mexv.z;
    h.w = (h.w - fm) * rstd * w.w + bb.w + mexv.w;
    *(float4*)(out_h + idx) = h;
  }
}

// ---------------- host ----------------
extern "C" void kernel_launch(void* const* d_in, const int* in_sizes, int n_in,
                              void* d_out, int out_size, void* d_ws, size_t ws_size,
                              hipStream_t stream) {
  const float* gx     = (const float*)d_in[0];
  const float* mx     = (const float*)d_in[1];
  const int*   me_i   = (const int*)  d_in[2];
  const float* me_x   = (const float*)d_in[3];
  const int*   g2me_i = (const int*)  d_in[4];
  const float* g2me_x = (const float*)d_in[5];
  const int*   m2ge_i = (const int*)  d_in[6];
  const float* m2ge_x = (const float*)d_in[7];
  const float* W1     = (const float*)d_in[8];
  const float* b1     = (const float*)d_in[9];
  const float* W2     = (const float*)d_in[10];
  const float* b2     = (const float*)d_in[11];
  const float* W3     = (const float*)d_in[12];
  const float* b3     = (const float*)d_in[13];
  const float* ln_w   = (const float*)d_in[14];
  const float* ln_b   = (const float*)d_in[15];

  const long n_gx  = GN_ * EMB, n_mx = MN_ * EMB, n_mei = 2 * ME_;
  const long n_mex = ME_ * EMB, n_gi = 2 * GE_,  n_gex = GE_ * EMB;

  float* out = (float*)d_out;
  float* o_gx  = out;
  float* o_mx  = o_gx + n_gx;
  float* o_mei = o_mx + n_mx;           // raw int bits, copied as-is
  float* o_mex = o_mei + n_mei;
  float* o_g2i = o_mex + n_mex;
  float* o_g2x = o_g2i + n_gi;
  float* o_m2i = o_g2x + n_gex;
  float* o_m2x = o_m2i + n_gi;

  // workspace: [0,16) f64 accumulators; [64,...) bf16-swizzled weights
  double*         accum = (double*)d_ws;
  unsigned short* w1s   = (unsigned short*)((char*)d_ws + 64);
  unsigned short* w2s   = w1s + (long)XDIM * H1D;
  unsigned short* w3s   = w2s + (long)H1D * H2D;

  init_accum_kernel<<<1, 1, 0, stream>>>(accum);
  convw_kernel<<<(XDIM * H1D + 255) / 256, 256, 0, stream>>>(W1, w1s, XDIM, H1D);
  convw_kernel<<<(H1D * H2D + 255) / 256, 256, 0, stream>>>(W2, w2s, H1D, H2D);
  convw_kernel<<<(H2D * (int)EMB + 255) / 256, 256, 0, stream>>>(W3, w3s, H2D, (int)EMB);

  const int tiles = (int)((ME_ + 15) / 16);
  mlp_kernel<<<tiles, 128, 0, stream>>>(mx, me_i, me_x, w1s, w2s, w3s,
                                        b1, b2, b3, o_mex, accum);

  const long nvec = n_mex / 4;
  norm_kernel<<<(int)((nvec + 255) / 256), 256, 0, stream>>>(o_mex, me_x, ln_w, ln_b, accum);

  // passthrough copies (graph-capture-safe async D2D)
  hipMemcpyAsync(o_gx,  gx,     n_gx  * 4, hipMemcpyDeviceToDevice, stream);
  hipMemcpyAsync(o_mx,  mx,     n_mx  * 4, hipMemcpyDeviceToDevice, stream);
  hipMemcpyAsync(o_mei, me_i,   n_mei * 4, hipMemcpyDeviceToDevice, stream);
  hipMemcpyAsync(o_g2i, g2me_i, n_gi  * 4, hipMemcpyDeviceToDevice, stream);
  hipMemcpyAsync(o_g2x, g2me_x, n_gex * 4, hipMemcpyDeviceToDevice, stream);
  hipMemcpyAsync(o_m2i, m2ge_i, n_gi  * 4, hipMemcpyDeviceToDevice, stream);
  hipMemcpyAsync(o_m2x, m2ge_x, n_gex * 4, hipMemcpyDeviceToDevice, stream);
}
